// HGMAE_20151986553169
// MI455X (gfx1250) — compile-verified, matrix operand order checked by reading, not run on hardware
//
#include <hip/hip_runtime.h>

// ---------------------------------------------------------------------------
// HGMAE GIN encoder/decoder for MI455X (gfx1250).
//
// Roofline: GEMMs ~0.6 TFLOP (negligible for the WMMA units); edge
// scatter/gather ~13 GB -> ~0.6 ms at 23.3 TB/s, with the 100 MB agg/h
// buffers resident in the 192 MB L2 so f32 atomics resolve in L2.
// GEMMs: v_wmma_f32_16x16x32_bf16, 128x128 block tile, LDS double-buffered,
// async global->LDS staging (ASYNCcnt), weights pre-swizzled fragment-major
// so fragment reads are ds_load_b128.
// ---------------------------------------------------------------------------

typedef unsigned short u16;
typedef unsigned int   u32;

typedef __attribute__((ext_vector_type(16))) __bf16 v16bf;
typedef __attribute__((ext_vector_type(8)))  float  v8f;
typedef __attribute__((__vector_size__(16))) int    v4i;

static constexpr int N_NODES = 50000;
static constexpr int N_EDGES = 500000;
static constexpr int EMB  = 500,  EMBP = 512;
static constexpr int HID  = 1000, HIDP = 1024;
static constexpr int OUTD = 119,  OUTP = 128;
static constexpr int LAYERS = 5;
static constexpr int MP = ((N_NODES + 127) / 128) * 128;   // 50048 padded rows
static constexpr float BN_EPS = 1e-5f;

#if __has_builtin(__builtin_amdgcn_global_load_async_to_lds_b128)
#define HAS_ASYNC_LDS 1
#endif

__device__ __forceinline__ u16 f2bf(float f) {
  u32 u = __builtin_bit_cast(u32, f);
  u32 r = u + 0x7FFFu + ((u >> 16) & 1u);   // round-to-nearest-even
  return (u16)(r >> 16);
}

union Frag16 {            // one WMMA 16-bit operand fragment (8 VGPRs)
  u32  u32v[8];
  uint4 q[2];
  v16bf v;
};

// 16-byte global -> LDS copy; async DMA path on CDNA5.
// Builtin signature (probe-verified): (global v4i*, lds v4i*, imm, imm)
__device__ __forceinline__ void async_copy16(const void* g, void* l) {
#ifdef HAS_ASYNC_LDS
  __builtin_amdgcn_global_load_async_to_lds_b128(
      (__attribute__((address_space(1))) v4i*)g,
      (__attribute__((address_space(3))) v4i*)l, 0, 0);
#else
  *(uint4*)l = *(const uint4*)g;
#endif
}

__device__ __forceinline__ void wait_async0() {
#ifdef HAS_ASYNC_LDS
#if __has_builtin(__builtin_amdgcn_s_wait_asynccnt)
  __builtin_amdgcn_s_wait_asynccnt(0);
#else
  asm volatile("s_wait_asynccnt 0x0" ::: "memory");
#endif
#endif
}

// ---------------------------------------------------------------------------
// Param prep kernels
// ---------------------------------------------------------------------------
// Swizzle weights into fragment-major layout:
//   O[kc][nt][lane][v] (u32 = packed bf16 pair) with
//   n = nt*16 + (lane&15), k = kc*32 + (v>>2)*16 + (lane>>4)*8 + (v&3)*2 (+j)
// so one lane's WMMA B-fragment is 32 contiguous bytes.
__global__ void convert_w_swz(const float* __restrict__ W, u32* __restrict__ O,
                              int K, int Nsrc, int Kp, int Np) {
  int idx = blockIdx.x * blockDim.x + threadIdx.x;
  int total = (Kp >> 5) * (Np >> 4) * 256;
  if (idx >= total) return;
  int v    = idx & 7;
  int lane = (idx >> 3) & 31;
  int nt   = (idx >> 8) % (Np >> 4);
  int kc   = (idx >> 8) / (Np >> 4);
  int n = nt * 16 + (lane & 15);
  int k = kc * 32 + ((v >> 2) << 4) + ((lane >> 4) << 3) + ((v & 3) << 1);
  float f0 = (k     < K && n < Nsrc) ? W[(size_t)k       * Nsrc + n] : 0.f;
  float f1 = (k + 1 < K && n < Nsrc) ? W[(size_t)(k + 1) * Nsrc + n] : 0.f;
  O[idx] = (u32)f2bf(f0) | ((u32)f2bf(f1) << 16);
}

__global__ void pad_vec(const float* __restrict__ src, float* __restrict__ dst,
                        int Nsrc, int Np) {
  int n = blockIdx.x * blockDim.x + threadIdx.x;
  if (n >= Np) return;
  dst[n] = (n < Nsrc) ? src[n] : 0.f;
}

__global__ void bn_prep(const float* __restrict__ g, const float* __restrict__ b,
                        const float* __restrict__ mean, const float* __restrict__ var,
                        float* __restrict__ scale, float* __restrict__ shift,
                        int Nsrc, int Np) {
  int n = blockIdx.x * blockDim.x + threadIdx.x;
  if (n >= Np) return;
  if (n < Nsrc) {
    float s = g[n] * rsqrtf(var[n] + BN_EPS);
    scale[n] = s;
    shift[n] = b[n] - mean[n] * s;
  } else {
    scale[n] = 0.f; shift[n] = 0.f;
  }
}

// h[m, :] = atom_emb1[x[m,0]] + atom_emb2[x[m,1]], zero pads (rows & cols)
__global__ void init_h(const int* __restrict__ x,
                       const float* __restrict__ e1, const float* __restrict__ e2,
                       float* __restrict__ h) {
  int idx = blockIdx.x * blockDim.x + threadIdx.x;
  if (idx >= MP * EMBP) return;
  int m = idx / EMBP, c = idx % EMBP;
  float v = 0.f;
  if (m < N_NODES && c < EMB)
    v = e1[(size_t)x[2 * m] * EMB + c] + e2[(size_t)x[2 * m + 1] * EMB + c];
  h[idx] = v;
}

// ---------------------------------------------------------------------------
// Scatter: agg[dst] += h[src] + ee1[ea0] + ee2[ea1]   (one wave per edge)
// agg rows are L2-resident -> no-return f32 atomics are the fast path.
// ---------------------------------------------------------------------------
__global__ __launch_bounds__(256) void scatter_k(
    const float* __restrict__ h,
    const float* __restrict__ ee1, const float* __restrict__ ee2,
    const int* __restrict__ edge_index, const int* __restrict__ edge_attr,
    float* __restrict__ agg) {
  int e = blockIdx.x * 8 + (threadIdx.x >> 5);
  if (e >= N_EDGES + N_NODES) return;
  int lane = threadIdx.x & 31;
  int s, d, a0, a1;
  if (e < N_EDGES) {
    s = edge_index[e];
    d = edge_index[N_EDGES + e];
    a0 = edge_attr[2 * e];
    a1 = edge_attr[2 * e + 1];
  } else {                         // self loop: token 4, dir 0
    s = d = e - N_EDGES;
    a0 = 4; a1 = 0;
  }
  const float4* hs = (const float4*)(h + (size_t)s * EMBP);
  const float4* p1 = (const float4*)(ee1 + (size_t)a0 * EMB);
  const float4* p2 = (const float4*)(ee2 + (size_t)a1 * EMB);
  float* ad = agg + (size_t)d * EMBP;
  for (int c = lane; c < EMB / 4; c += 32) {   // 125 float4 chunks
    float4 hv = hs[c], v1 = p1[c], v2 = p2[c];
    unsafeAtomicAdd(&ad[4 * c + 0], hv.x + v1.x + v2.x);
    unsafeAtomicAdd(&ad[4 * c + 1], hv.y + v1.y + v2.y);
    unsafeAtomicAdd(&ad[4 * c + 2], hv.z + v1.z + v2.z);
    unsafeAtomicAdd(&ad[4 * c + 3], hv.w + v1.w + v2.w);
  }
}

// ---------------------------------------------------------------------------
// WMMA GEMM: C = epilogue(A @ B)
//  AMODE 0: A fp32 | 1: A fp32 + PReLU on load | 2: A bf16 (raw async copy)
//  EPI   0: relu(x+bias) -> bf16 tmp
//        1: bn(x+bias) (+opt relu) -> f32
//        2: mask-zero -> f32
//        3: x+bias -> f32, stores guarded to n<n_limit (final 119-wide out)
// Block 256 thr = 8 waves; block tile 128x128, K-step 32; wave owns 32x64.
// LDS double-buffered; B comes pre-swizzled (fragment-major).
// ---------------------------------------------------------------------------
template <int EPI>
__device__ __forceinline__ void epi_store(
    v8f acc, int mb, int nb, void* Cv, int ldc, int M,
    const float* bias, const float* bnscale, const float* bnshift,
    const int* mask, int relu_flag, int n_limit) {
  int n = nb;
#pragma unroll
  for (int r = 0; r < 8; ++r) {
    int m = mb + r;
    if (m >= M) continue;
    float val = acc[r];
    if (EPI == 0) {
      float x = val + bias[n];
      x = x > 0.f ? x : 0.f;
      ((u16*)Cv)[(size_t)m * ldc + n] = f2bf(x);
    } else if (EPI == 1) {
      float x = (val + bias[n]) * bnscale[n] + bnshift[n];
      if (relu_flag) x = x > 0.f ? x : 0.f;
      ((float*)Cv)[(size_t)m * ldc + n] = x;
    } else if (EPI == 2) {
      ((float*)Cv)[(size_t)m * ldc + n] = mask[m] ? 0.f : val;
    } else {
      if (n < n_limit)
        ((float*)Cv)[(size_t)m * ldc + n] = val + bias[n];
    }
  }
}

template <int AMODE, int EPI>
__global__ __launch_bounds__(256) void gemm_k(
    const void* __restrict__ Av, int lda,
    const u32* __restrict__ B, int ldb,         // B pre-swizzled; ldb = Np
    void* __restrict__ Cv, int ldc,
    int M, int Kp,
    const float* __restrict__ bias,
    const float* __restrict__ bnscale, const float* __restrict__ bnshift,
    const int* __restrict__ mask, const float* __restrict__ prelu_a,
    int relu_flag, int n_limit) {
  __shared__ __attribute__((aligned(16))) u16 As[2][128 * 40];  // 128r x 32k (+pad)
  __shared__ __attribute__((aligned(16))) u32 Bs[2][2048];      // 8 ntiles x 256

  const int t = threadIdx.x;
  const int lane = t & 31, wave = t >> 5;
  const int wm = wave & 3, wn = wave >> 2;         // 4 (M) x 2 (N) wave grid
  const int half = lane >> 4, ml = lane & 15;
  const int m0 = blockIdx.y * 128;
  const int n0 = blockIdx.x * 128;
  const int ntTot = ldb >> 4;
  const float pa = (AMODE == 1) ? prelu_a[0] : 0.f;

  v8f acc[2][4] = {};

  auto stageA = [&](int k0, int buf) {
    if constexpr (AMODE <= 1) {
      // fp32 -> bf16 conversion (pad rows are valid zeroed memory: no guard).
      // Loads split from converts so all 4 are in flight before any wait.
      const float* A = (const float*)Av;
      int r = t >> 3, c4 = (t & 7) << 2;
      float4 vv[4];
#pragma unroll
      for (int it = 0; it < 4; ++it)
        vv[it] = *(const float4*)(A + (size_t)(m0 + r + it * 32) * lda + k0 + c4);
#pragma unroll
      for (int it = 0; it < 4; ++it) {
        float4 v = vv[it];
        if (AMODE == 1) {
          v.x = v.x >= 0.f ? v.x : pa * v.x;
          v.y = v.y >= 0.f ? v.y : pa * v.y;
          v.z = v.z >= 0.f ? v.z : pa * v.z;
          v.w = v.w >= 0.f ? v.w : pa * v.w;
        }
        u16* d = &As[buf][(r + it * 32) * 40 + c4];
        d[0] = f2bf(v.x); d[1] = f2bf(v.y); d[2] = f2bf(v.z); d[3] = f2bf(v.w);
      }
    } else {
      // raw bf16 copy (pad rows only feed discarded output rows)
      const u16* A = (const u16*)Av;
#pragma unroll
      for (int it = 0; it < 2; ++it) {
        int c = t + it * 256;
        int r = c >> 2, part = c & 3;
        async_copy16(A + (size_t)(m0 + r) * lda + k0 + part * 8,
                     &As[buf][r * 40 + part * 8]);
      }
    }
  };
  auto stageB = [&](int k0, int buf) {
    // contiguous 8 KB chunk: 8 ntiles of this block's 128 columns
    const u32* Bg = B + ((size_t)(k0 >> 5) * ntTot + (n0 >> 4)) * 256;
#pragma unroll
    for (int it = 0; it < 2; ++it)
      async_copy16(Bg + (t + it * 256) * 4, &Bs[buf][(t + it * 256) * 4]);
  };

  stageA(0, 0);
  stageB(0, 0);

  for (int k0 = 0; k0 < Kp; k0 += 32) {
    const int buf = (k0 >> 5) & 1;
    wait_async0();        // current buffer's async copies done (this wave)
    __syncthreads();      // all waves' tiles visible
    if (k0 + 32 < Kp) {   // prefetch next tile into the other buffer
      stageA(k0 + 32, buf ^ 1);
      stageB(k0 + 32, buf ^ 1);
    }

    // ---- fragments: 2x ds_load_b128 each ----
    Frag16 af[2], bf[4];
    const u16* Ab = As[buf];
#pragma unroll
    for (int i = 0; i < 2; ++i) {
      const int am = (wm * 32 + i * 16 + ml) * 40;
      af[i].q[0] = *(const uint4*)&Ab[am + (half << 3)];
      af[i].q[1] = *(const uint4*)&Ab[am + 16 + (half << 3)];
    }
    const u32* Bb = Bs[buf];
#pragma unroll
    for (int j = 0; j < 4; ++j) {
      const int bt = (wn * 4 + j) * 256 + lane * 8;
      bf[j].q[0] = *(const uint4*)&Bb[bt];
      bf[j].q[1] = *(const uint4*)&Bb[bt + 4];
    }
#pragma unroll
    for (int i = 0; i < 2; ++i)
#pragma unroll
      for (int j = 0; j < 4; ++j)
        acc[i][j] = __builtin_amdgcn_wmma_f32_16x16x32_bf16(
            false, af[i].v, false, bf[j].v, (short)0, acc[i][j], false, false);
  }

  const int mb0 = m0 + wm * 32 + half * 8;
  const int nb0 = n0 + wn * 64 + ml;
#pragma unroll
  for (int i = 0; i < 2; ++i)
#pragma unroll
    for (int j = 0; j < 4; ++j)
      epi_store<EPI>(acc[i][j], mb0 + i * 16, nb0 + j * 16, Cv, ldc, M,
                     bias, bnscale, bnshift, mask, relu_flag, n_limit);
}

// ---------------------------------------------------------------------------
// Host-side orchestration
// ---------------------------------------------------------------------------
extern "C" void kernel_launch(void* const* d_in, const int* in_sizes, int n_in,
                              void* d_out, int out_size, void* d_ws, size_t ws_size,
                              hipStream_t stream) {
  const int*   x          = (const int*)  d_in[0];
  const int*   edge_index = (const int*)  d_in[1];
  const int*   edge_attr  = (const int*)  d_in[2];
  const int*   mask       = (const int*)  d_in[3];   // bool mask (nonzero = masked)
  const float* atom_emb1  = (const float*)d_in[4];
  const float* atom_emb2  = (const float*)d_in[5];
  const float* enc_eemb1  = (const float*)d_in[6];   // [5,6,500]
  const float* enc_eemb2  = (const float*)d_in[7];   // [5,3,500]
  const float* enc_W1     = (const float*)d_in[8];   // [5,500,1000]
  const float* enc_b1     = (const float*)d_in[9];   // [5,1000]
  const float* enc_W2     = (const float*)d_in[10];  // [5,1000,500]
  const float* enc_b2     = (const float*)d_in[11];  // [5,500]
  const float* bn_gamma   = (const float*)d_in[12];
  const float* bn_beta    = (const float*)d_in[13];
  const float* bn_mean    = (const float*)d_in[14];
  const float* bn_var     = (const float*)d_in[15];
  const float* prelu_a    = (const float*)d_in[16];
  const float* n2d_W      = (const float*)d_in[17];  // [500,500]
  const float* dec_eemb1  = (const float*)d_in[18];  // [6,500]
  const float* dec_eemb2  = (const float*)d_in[19];  // [3,500]
  const float* dec_W1     = (const float*)d_in[20];  // [500,1000]
  const float* dec_b1     = (const float*)d_in[21];  // [1000]
  const float* dec_W2     = (const float*)d_in[22];  // [1000,119]
  const float* dec_b2     = (const float*)d_in[23];  // [119]

  // ---- carve workspace ----
  size_t off = 0;
  auto carve = [&](size_t bytes) {
    void* p = (char*)d_ws + off;
    off = (off + bytes + 255) & ~(size_t)255;
    return p;
  };
  float* h    = (float*)carve((size_t)MP * EMBP * 4);
  float* agg  = (float*)carve((size_t)MP * EMBP * 4);
  u16*   tmp  = (u16*)  carve((size_t)MP * HIDP * 2);
  float* h2   = (float*)tmp;   // alias: same byte size, lifetimes disjoint
  u32* wenc1  = (u32*)carve((size_t)LAYERS * EMBP * HIDP / 2 * 4);
  u32* wenc2  = (u32*)carve((size_t)LAYERS * HIDP * EMBP / 2 * 4);
  u32* wn2d   = (u32*)carve((size_t)EMBP * EMBP / 2 * 4);
  u32* wdec1  = (u32*)carve((size_t)EMBP * HIDP / 2 * 4);
  u32* wdec2  = (u32*)carve((size_t)HIDP * OUTP / 2 * 4);
  float* benc1 = (float*)carve((size_t)LAYERS * HIDP * 4);
  float* benc2 = (float*)carve((size_t)LAYERS * EMBP * 4);
  float* bnsc  = (float*)carve((size_t)LAYERS * EMBP * 4);
  float* bnsh  = (float*)carve((size_t)LAYERS * EMBP * 4);
  float* bdec1 = (float*)carve((size_t)HIDP * 4);
  float* bdec2 = (float*)carve((size_t)OUTP * 4);
  (void)ws_size; (void)n_in; (void)in_sizes; (void)out_size;

  const int TB = 256;
  auto blks = [](long n, int tb) { return (unsigned)((n + tb - 1) / tb); };

  // ---- param conversion (deterministic, re-run every call) ----
  for (int l = 0; l < LAYERS; ++l) {
    convert_w_swz<<<blks((long)EMBP * HIDP / 2, TB), TB, 0, stream>>>(
        enc_W1 + (size_t)l * EMB * HID, wenc1 + (size_t)l * EMBP * HIDP / 2, EMB, HID, EMBP, HIDP);
    convert_w_swz<<<blks((long)HIDP * EMBP / 2, TB), TB, 0, stream>>>(
        enc_W2 + (size_t)l * HID * EMB, wenc2 + (size_t)l * HIDP * EMBP / 2, HID, EMB, HIDP, EMBP);
    pad_vec<<<blks(HIDP, TB), TB, 0, stream>>>(enc_b1 + (size_t)l * HID, benc1 + (size_t)l * HIDP, HID, HIDP);
    pad_vec<<<blks(EMBP, TB), TB, 0, stream>>>(enc_b2 + (size_t)l * EMB, benc2 + (size_t)l * EMBP, EMB, EMBP);
    bn_prep<<<blks(EMBP, TB), TB, 0, stream>>>(
        bn_gamma + (size_t)l * EMB, bn_beta + (size_t)l * EMB,
        bn_mean + (size_t)l * EMB, bn_var + (size_t)l * EMB,
        bnsc + (size_t)l * EMBP, bnsh + (size_t)l * EMBP, EMB, EMBP);
  }
  convert_w_swz<<<blks((long)EMBP * EMBP / 2, TB), TB, 0, stream>>>(n2d_W, wn2d, EMB, EMB, EMBP, EMBP);
  convert_w_swz<<<blks((long)EMBP * HIDP / 2, TB), TB, 0, stream>>>(dec_W1, wdec1, EMB, HID, EMBP, HIDP);
  convert_w_swz<<<blks((long)HIDP * OUTP / 2, TB), TB, 0, stream>>>(dec_W2, wdec2, HID, OUTD, HIDP, OUTP);
  pad_vec<<<blks(HIDP, TB), TB, 0, stream>>>(dec_b1, bdec1, HID, HIDP);
  pad_vec<<<blks(OUTP, TB), TB, 0, stream>>>(dec_b2, bdec2, OUTD, OUTP);

  // ---- node embedding init ----
  init_h<<<blks((long)MP * EMBP, TB), TB, 0, stream>>>(x, atom_emb1, atom_emb2, h);

  const unsigned mBlk = (N_NODES + 127) / 128;           // 391
  dim3 g1(HIDP / 128, mBlk);                             // K=512  -> N=1024
  dim3 g2(EMBP / 128, mBlk);                             // K=1024 -> N=512
  dim3 g3(OUTP / 128, mBlk);                             // K=1024 -> N=128
  const unsigned edgeBlks = (N_EDGES + N_NODES + 7) / 8; // wave per edge

  // ---- encoder ----
  for (int l = 0; l < LAYERS; ++l) {
    (void)hipMemsetAsync(agg, 0, (size_t)MP * EMBP * 4, stream);
    scatter_k<<<edgeBlks, TB, 0, stream>>>(
        h, enc_eemb1 + (size_t)l * 6 * EMB, enc_eemb2 + (size_t)l * 3 * EMB,
        edge_index, edge_attr, agg);
    gemm_k<0, 0><<<g1, TB, 0, stream>>>(
        agg, EMBP, wenc1 + (size_t)l * EMBP * HIDP / 2, HIDP, tmp, HIDP,
        N_NODES, EMBP, benc1 + (size_t)l * HIDP,
        nullptr, nullptr, nullptr, nullptr, 0, 0);
    gemm_k<2, 1><<<g2, TB, 0, stream>>>(
        tmp, HIDP, wenc2 + (size_t)l * HIDP * EMBP / 2, EMBP, h, EMBP,
        N_NODES, HIDP, benc2 + (size_t)l * EMBP,
        bnsc + (size_t)l * EMBP, bnsh + (size_t)l * EMBP,
        nullptr, nullptr, (l < LAYERS - 1) ? 1 : 0, 0);
  }

  // ---- encoder_to_decoder: PReLU (on A load) @ n2d_W, then mask-zero ----
  gemm_k<1, 2><<<g2, TB, 0, stream>>>(
      h, EMBP, wn2d, EMBP, h2, EMBP, N_NODES, EMBP,
      nullptr, nullptr, nullptr, mask, prelu_a, 0, 0);

  // ---- decoder GIN ----
  (void)hipMemsetAsync(agg, 0, (size_t)MP * EMBP * 4, stream);
  scatter_k<<<edgeBlks, TB, 0, stream>>>(h2, dec_eemb1, dec_eemb2,
                                         edge_index, edge_attr, agg);
  gemm_k<0, 0><<<g1, TB, 0, stream>>>(           // overwrites tmp (h2 now dead)
      agg, EMBP, wdec1, HIDP, tmp, HIDP, N_NODES, EMBP, bdec1,
      nullptr, nullptr, nullptr, nullptr, 0, 0);
  gemm_k<2, 3><<<g3, TB, 0, stream>>>(
      tmp, HIDP, wdec2, OUTP, d_out, OUTD, N_NODES, HIDP, bdec2,
      nullptr, nullptr, nullptr, nullptr, 0, OUTD);
}
